// SSMoverGOOMs_84902913507664
// MI455X (gfx1250) — matrix operand order repaired and checked by reading, not compiled, not run on hardware
//
#include <hip/hip_runtime.h>
#include <hip/hip_bf16.h>

typedef _Float16 h8   __attribute__((ext_vector_type(8)));
typedef _Float16 v16h __attribute__((ext_vector_type(16)));
typedef float    v8f  __attribute__((ext_vector_type(8)));
typedef float    v2f  __attribute__((ext_vector_type(2)));

namespace {
constexpr int kBsz = 8, kLen = 1024, kDin = 512, kNh = 4, kDh = 64, kDout = 512;
constexpr int kTok    = kBsz * kLen;      // 8192 tokens
constexpr int kNst    = kNh * kDh;        // 256 state width
constexpr int kChunk  = 32;               // timesteps per scan chunk
constexpr int kNChunk = kLen / kChunk;    // 32
constexpr int kNSeq   = kBsz * kNh;       // 32 independent sequences
}

// ---------------------------------------------------------------------------
// WMMA fragment helpers (wave32 layouts per CDNA5 ISA 7.12.2)
__device__ __forceinline__ v16h load_afrag(const _Float16* p) {
    // A 16x32 f16: halves 0..15 -> K kb..kb+15 (contiguous)
    h8 lo = *(const h8*)p;
    h8 hi = *(const h8*)(p + 8);
    return __builtin_shufflevector(lo, hi, 0,1,2,3,4,5,6,7,8,9,10,11,12,13,14,15);
}
__device__ __forceinline__ v16h load_bfrag(const _Float16* p) {
    // B 32x16 f16 from N-major weights: halves 0..15 -> K kblk..kblk+15 (contiguous)
    h8 lo = *(const h8*)p;
    h8 hi = *(const h8*)(p + 8);
    return __builtin_shufflevector(lo, hi, 0,1,2,3,4,5,6,7,8,9,10,11,12,13,14,15);
}
__device__ __forceinline__ v8f wmma_f16(v16h a, v16h b, v8f c) {
    return __builtin_amdgcn_wmma_f32_16x16x32_f16(false, a, false, b, (short)0, c, false, false);
}
__device__ __forceinline__ void store_ctile(float* C, size_t ldc, int m0, int n0,
                                            int lane, v8f acc) {
    // C/D 16x16 f32: lane -> col, VGPR r -> row r (lanes 0-15) / row r+8 (lanes 16-31)
    int r15 = lane & 15, rb = (lane >> 4) << 3;
    float* po = C + (size_t)(m0 + rb) * ldc + n0 + r15;
#pragma unroll
    for (int r = 0; r < 8; ++r) po[(size_t)r * ldc] = acc[r];
}

// ---------------------------------------------------------------------------
// fp32 -> f16 elementwise convert
__global__ void cvt_f32_f16(const float* __restrict__ in, _Float16* __restrict__ out, int n) {
    int i = blockIdx.x * blockDim.x + threadIdx.x;
    if (i < n) out[i] = (_Float16)in[i];
}

// fp32 [rows,cols] -> f16 transposed [cols,rows] (makes WMMA B-fragment loads contiguous)
__global__ void transpose_f32_f16(const float* __restrict__ in, _Float16* __restrict__ out,
                                  int rows, int cols) {
    int i = blockIdx.x * blockDim.x + threadIdx.x;
    if (i >= rows * cols) return;
    int r = i / cols, c = i - r * cols;
    out[c * rows + r] = (_Float16)in[i];
}

// ---------------------------------------------------------------------------
// f16 WMMA GEMM, 32x32 register tile per wave (4 accumulators, fragments reused
// 2x each): C[M,N] = A[M,K] @ Bt[N,K]^T, fp32 accumulate. 2 b128 loads per WMMA.
__global__ void gemm_f16_wmma(const _Float16* __restrict__ A, const _Float16* __restrict__ Bt,
                              float* __restrict__ C, int M, int N, int K) {
    int lane = threadIdx.x & 31;
    int wid  = threadIdx.x >> 5;
    int tilesN = N >> 5;
    int tile = blockIdx.x * (int)(blockDim.x >> 5) + wid;
    if (tile >= (M >> 5) * tilesN) return;          // uniform per-wave; EXEC stays all-1
    int m0 = (tile / tilesN) << 5;
    int n0 = (tile % tilesN) << 5;
    int r15 = lane & 15, hi = lane >> 4;

    const _Float16* pa0 = A  + (size_t)(m0 + r15) * K + hi * 16;
    const _Float16* pa1 = pa0 + (size_t)16 * K;
    const _Float16* pb0 = Bt + (size_t)(n0 + r15) * K + hi * 16;
    const _Float16* pb1 = pb0 + (size_t)16 * K;

    v8f acc00 = {}, acc01 = {}, acc10 = {}, acc11 = {};
    for (int k = 0; k < K; k += 32) {
        v16h a0 = load_afrag(pa0 + k);
        v16h a1 = load_afrag(pa1 + k);
        v16h b0 = load_bfrag(pb0 + k);
        v16h b1 = load_bfrag(pb1 + k);
        acc00 = wmma_f16(a0, b0, acc00);
        acc01 = wmma_f16(a0, b1, acc01);
        acc10 = wmma_f16(a1, b0, acc10);
        acc11 = wmma_f16(a1, b1, acc11);
    }
    store_ctile(C, N, m0,      n0,      lane, acc00);
    store_ctile(C, N, m0,      n0 + 16, lane, acc01);
    store_ctile(C, N, m0 + 16, n0,      lane, acc10);
    store_ctile(C, N, m0 + 16, n0 + 16, lane, acc11);
}

// ---------------------------------------------------------------------------
// Fused output GEMM, 32x32 tile/wave: Y[8192,512] = X[8192,256]@Ct^T + U[8192,512]@Dt^T
__global__ void gemm_out_wmma(const _Float16* __restrict__ X,  const _Float16* __restrict__ Ct,
                              const _Float16* __restrict__ U,  const _Float16* __restrict__ Dt,
                              float* __restrict__ Y) {
    int lane = threadIdx.x & 31;
    int wid  = threadIdx.x >> 5;
    int tilesN = kDout >> 5;
    int tile = blockIdx.x * (int)(blockDim.x >> 5) + wid;
    if (tile >= (kTok >> 5) * tilesN) return;
    int m0 = (tile / tilesN) << 5;
    int n0 = (tile % tilesN) << 5;
    int r15 = lane & 15, hi = lane >> 4;

    v8f acc00 = {}, acc01 = {}, acc10 = {}, acc11 = {};
    {   // X @ C  (K = 256)
        const _Float16* pa0 = X  + (size_t)(m0 + r15) * kNst + hi * 16;
        const _Float16* pa1 = pa0 + (size_t)16 * kNst;
        const _Float16* pb0 = Ct + (size_t)(n0 + r15) * kNst + hi * 16;
        const _Float16* pb1 = pb0 + (size_t)16 * kNst;
        for (int k = 0; k < kNst; k += 32) {
            v16h a0 = load_afrag(pa0 + k);
            v16h a1 = load_afrag(pa1 + k);
            v16h b0 = load_bfrag(pb0 + k);
            v16h b1 = load_bfrag(pb1 + k);
            acc00 = wmma_f16(a0, b0, acc00);
            acc01 = wmma_f16(a0, b1, acc01);
            acc10 = wmma_f16(a1, b0, acc10);
            acc11 = wmma_f16(a1, b1, acc11);
        }
    }
    {   // U @ D  (K = 512)
        const _Float16* pa0 = U  + (size_t)(m0 + r15) * kDin + hi * 16;
        const _Float16* pa1 = pa0 + (size_t)16 * kDin;
        const _Float16* pb0 = Dt + (size_t)(n0 + r15) * kDin + hi * 16;
        const _Float16* pb1 = pb0 + (size_t)16 * kDin;
        for (int k = 0; k < kDin; k += 32) {
            v16h a0 = load_afrag(pa0 + k);
            v16h a1 = load_afrag(pa1 + k);
            v16h b0 = load_bfrag(pb0 + k);
            v16h b1 = load_bfrag(pb1 + k);
            acc00 = wmma_f16(a0, b0, acc00);
            acc01 = wmma_f16(a0, b1, acc01);
            acc10 = wmma_f16(a1, b0, acc10);
            acc11 = wmma_f16(a1, b1, acc11);
        }
    }
    store_ctile(Y, kDout, m0,      n0,      lane, acc00);
    store_ctile(Y, kDout, m0,      n0 + 16, lane, acc01);
    store_ctile(Y, kDout, m0 + 16, n0,      lane, acc10);
    store_ctile(Y, kDout, m0 + 16, n0 + 16, lane, acc11);
}

// ---------------------------------------------------------------------------
// 64x64 fp32 matrix square (Out = Xm @ Xm) for the A-power chain.
// One block, 16 waves, each a 16x16 tile via v_wmma_f32_16x16x4_f32.
__global__ void gemm64_f32_wmma(const float* __restrict__ Xm, float* __restrict__ Out) {
    int lane = threadIdx.x & 31;
    int wid  = threadIdx.x >> 5;                    // 0..15
    int m0 = (wid >> 2) << 4, n0 = (wid & 3) << 4;
    int r15 = lane & 15;
#if __has_builtin(__builtin_amdgcn_wmma_f32_16x16x4_f32)
    int koff = (lane >> 4) << 1;                    // f32 A/B frag: K = koff, koff+1
    v8f acc = {};
    for (int k = 0; k < 64; k += 4) {
        v2f a, b;
        a.x = Xm[(m0 + r15) * 64 + k + koff];
        a.y = Xm[(m0 + r15) * 64 + k + koff + 1];
        b.x = Xm[(k + koff) * 64 + n0 + r15];
        b.y = Xm[(k + koff + 1) * 64 + n0 + r15];
        acc = __builtin_amdgcn_wmma_f32_16x16x4_f32(false, a, false, b, (short)0, acc, false, false);
    }
    int rb = (lane >> 4) << 3;
    float* po = Out + (m0 + rb) * 64 + n0 + r15;
#pragma unroll
    for (int r = 0; r < 8; ++r) po[r * 64] = acc[r];
#else
    int rb = (lane >> 4) << 3;
    for (int r = 0; r < 8; ++r) {
        float s = 0.f;
        for (int k = 0; k < 64; ++k)
            s += Xm[(m0 + rb + r) * 64 + k] * Xm[k * 64 + n0 + r15];
        Out[(m0 + rb + r) * 64 + n0 + r15] = s;
    }
#endif
}

// ---------------------------------------------------------------------------
// Scan phase 1: per (seq, chunk), local recurrence x_t = x_{t-1} @ A + Bu_t
// with x = 0 init; store chunk-end state. 1 wave/block, A cached in LDS.
__global__ void scan_local(const float* __restrict__ Bu, const float* __restrict__ Amat,
                           float* __restrict__ chunkEnd) {
    __shared__ float As[64 * 64];
    __shared__ float xs[64];
    int lane  = threadIdx.x;                        // 0..31
    int chunk = blockIdx.x;                         // 0..kNChunk-1
    int seq   = blockIdx.y;                         // 0..kNSeq-1
    int batch = seq >> 2, head = seq & 3;
    const float4* A4 = (const float4*)Amat;
    float4* As4 = (float4*)As;
    for (int i = lane; i < 1024; i += 32) As4[i] = A4[i];
    xs[lane] = 0.f; xs[lane + 32] = 0.f;
    __syncthreads();
    int tok0 = batch * kLen + chunk * kChunk;
    for (int s = 0; s < kChunk; ++s) {
        const float* bu = Bu + (size_t)(tok0 + s) * kNst + head * kDh;
        float a0 = bu[lane], a1 = bu[lane + 32];
#pragma unroll 8
        for (int i = 0; i < 64; ++i) {
            float xi = xs[i];
            a0 = fmaf(xi, As[i * 64 + lane], a0);
            a1 = fmaf(xi, As[i * 64 + lane + 32], a1);
        }
        __syncthreads();
        xs[lane] = a0; xs[lane + 32] = a1;
        __syncthreads();
    }
    float* e = chunkEnd + (size_t)(seq * kNChunk + chunk) * 64;
    e[lane] = xs[lane]; e[lane + 32] = xs[lane + 32];
}

// Scan phase 1.5: per-sequence prefix over chunks using A^kChunk:
// s_{c+1} = s_c @ A32 + end_c, s_0 = init_states[head]. 1 wave/block.
__global__ void scan_prefix(const float* __restrict__ chunkEnd, const float* __restrict__ Apow,
                            const float* __restrict__ init_states, float* __restrict__ chunkStart) {
    __shared__ float As[64 * 64];
    __shared__ float xs[64];
    int lane = threadIdx.x;
    int seq = blockIdx.x;
    int head = seq & 3;
    const float4* A4 = (const float4*)Apow;
    float4* As4 = (float4*)As;
    for (int i = lane; i < 1024; i += 32) As4[i] = A4[i];
    xs[lane]      = init_states[head * 64 + lane];
    xs[lane + 32] = init_states[head * 64 + lane + 32];
    __syncthreads();
    float* s0 = chunkStart + (size_t)seq * kNChunk * 64;
    s0[lane] = xs[lane]; s0[lane + 32] = xs[lane + 32];
    for (int c = 0; c < kNChunk - 1; ++c) {
        const float* e = chunkEnd + (size_t)(seq * kNChunk + c) * 64;
        float a0 = e[lane], a1 = e[lane + 32];
#pragma unroll 8
        for (int i = 0; i < 64; ++i) {
            float xi = xs[i];
            a0 = fmaf(xi, As[i * 64 + lane], a0);
            a1 = fmaf(xi, As[i * 64 + lane + 32], a1);
        }
        __syncthreads();
        xs[lane] = a0; xs[lane + 32] = a1;
        __syncthreads();
        float* sc = chunkStart + (size_t)(seq * kNChunk + c + 1) * 64;
        sc[lane] = xs[lane]; sc[lane + 32] = xs[lane + 32];
    }
}

// Scan phase 2: re-run recurrence with correct chunk-start state; fuse the GOOM
// normalization x * e^2 / max_j|x_j| and emit f16 X for the output GEMM.
__global__ void scan_final(const float* __restrict__ Bu, const float* __restrict__ Amat,
                           const float* __restrict__ chunkStart, _Float16* __restrict__ Xf16) {
    __shared__ float As[64 * 64];
    __shared__ float xs[64];
    int lane  = threadIdx.x;
    int chunk = blockIdx.x;
    int seq   = blockIdx.y;
    int batch = seq >> 2, head = seq & 3;
    const float4* A4 = (const float4*)Amat;
    float4* As4 = (float4*)As;
    for (int i = lane; i < 1024; i += 32) As4[i] = A4[i];
    const float* s0 = chunkStart + (size_t)(seq * kNChunk + chunk) * 64;
    xs[lane] = s0[lane]; xs[lane + 32] = s0[lane + 32];
    __syncthreads();
    int tok0 = batch * kLen + chunk * kChunk;
    for (int s = 0; s < kChunk; ++s) {
        const float* bu = Bu + (size_t)(tok0 + s) * kNst + head * kDh;
        float a0 = bu[lane], a1 = bu[lane + 32];
#pragma unroll 8
        for (int i = 0; i < 64; ++i) {
            float xi = xs[i];
            a0 = fmaf(xi, As[i * 64 + lane], a0);
            a1 = fmaf(xi, As[i * 64 + lane + 32], a1);
        }
        // exp(log_x - max + 2).real  ==  x * e^2 / max|x|   (per head, per step)
        float m = fmaxf(fabsf(a0), fabsf(a1));
        for (int off = 16; off > 0; off >>= 1)
            m = fmaxf(m, __shfl_xor(m, off, 32));
        float sc = 7.3890560989306495f / m;         // e^2 / max
        _Float16* xo = Xf16 + (size_t)(tok0 + s) * kNst + head * kDh;
        xo[lane]      = (_Float16)(a0 * sc);
        xo[lane + 32] = (_Float16)(a1 * sc);
        __syncthreads();
        xs[lane] = a0; xs[lane + 32] = a1;
        __syncthreads();
    }
}

// ---------------------------------------------------------------------------
extern "C" void kernel_launch(void* const* d_in, const int* in_sizes, int n_in,
                              void* d_out, int out_size, void* d_ws, size_t ws_size,
                              hipStream_t stream) {
    const float* u    = (const float*)d_in[0];   // [8,1024,512]
    const float* init = (const float*)d_in[1];   // [4,64]
    const float* A    = (const float*)d_in[2];   // [64,64]
    const float* B    = (const float*)d_in[3];   // [512,256]
    const float* C    = (const float*)d_in[4];   // [256,512]
    const float* D    = (const float*)d_in[5];   // [512,512]
    float* y = (float*)d_out;                    // [8,1024,512]

    char* ws = (char*)d_ws;
    size_t off = 0;
    auto alloc = [&](size_t bytes) -> void* {
        void* p = ws + off;
        off += (bytes + 255) & ~(size_t)255;
        return p;
    };
    _Float16* uf16  = (_Float16*)alloc((size_t)kTok * kDin * 2);   // 8 MB
    _Float16* Btf   = (_Float16*)alloc((size_t)kNst * kDin * 2);   // Bt [256,512]
    _Float16* Ctf   = (_Float16*)alloc((size_t)kDout * kNst * 2);  // Ct [512,256]
    _Float16* Dtf   = (_Float16*)alloc((size_t)kDout * kDin * 2);  // Dt [512,512]
    float*    BuBuf = (float*)alloc((size_t)kTok * kNst * 4);      // 8 MB
    float*    P1    = (float*)alloc(64 * 64 * 4);
    float*    P2    = (float*)alloc(64 * 64 * 4);
    float*    cEnd  = (float*)alloc((size_t)kNSeq * kNChunk * 64 * 4);
    float*    cStart= (float*)alloc((size_t)kNSeq * kNChunk * 64 * 4);
    _Float16* Xf16  = (_Float16*)alloc((size_t)kTok * kNst * 2);   // 4 MB

    // 1) precision conversion + weight pre-transposition (contiguous B-fragments)
    cvt_f32_f16<<<(kTok * kDin + 255) / 256, 256, 0, stream>>>(u, uf16, kTok * kDin);
    transpose_f32_f16<<<(kDin * kNst + 255) / 256, 256, 0, stream>>>(B, Btf, kDin, kNst);
    transpose_f32_f16<<<(kNst * kDout + 255) / 256, 256, 0, stream>>>(C, Ctf, kNst, kDout);
    transpose_f32_f16<<<(kDin * kDout + 255) / 256, 256, 0, stream>>>(D, Dtf, kDin, kDout);

    // 2) Bu = u @ B   (f16 WMMA, 32x32 reg tile/wave) : 2048 tiles / 8 waves = 256 blocks
    gemm_f16_wmma<<<(kTok / 32) * (kNst / 32) / 8, 256, 0, stream>>>(
        uf16, Btf, BuBuf, kTok, kNst, kDin);

    // 3) A^32 by repeated squaring (fp32 WMMA): A->A^2->A^4->A^8->A^16->A^32
    gemm64_f32_wmma<<<1, 512, 0, stream>>>(A,  P1);   // A^2
    gemm64_f32_wmma<<<1, 512, 0, stream>>>(P1, P2);   // A^4
    gemm64_f32_wmma<<<1, 512, 0, stream>>>(P2, P1);   // A^8
    gemm64_f32_wmma<<<1, 512, 0, stream>>>(P1, P2);   // A^16
    gemm64_f32_wmma<<<1, 512, 0, stream>>>(P2, P1);   // A^32 -> P1

    // 4) chunked scan: local scans || -> per-seq prefix -> final scan + GOOM norm
    scan_local<<<dim3(kNChunk, kNSeq), 32, 0, stream>>>(BuBuf, A, cEnd);
    scan_prefix<<<kNSeq, 32, 0, stream>>>(cEnd, P1, init, cStart);
    scan_final<<<dim3(kNChunk, kNSeq), 32, 0, stream>>>(BuBuf, A, cStart, Xf16);

    // 5) y = X @ C + u @ D  (fused, 32x32 reg tile/wave) : 4096 tiles / 8 = 512 blocks
    gemm_out_wmma<<<(kTok / 32) * (kDout / 32) / 8, 256, 0, stream>>>(
        Xf16, Ctf, uf16, Dtf, y);
}
